// Multihead_Attention_53807350284483
// MI455X (gfx1250) — compile-verified, matrix operand order checked by reading, not run on hardware
//
#include <hip/hip_runtime.h>

typedef __bf16 bf16_t;
typedef __attribute__((ext_vector_type(16))) __bf16 v16bf;
typedef __attribute__((ext_vector_type(8)))  float   v8f;

#define OUT_BF16     0
#define OUT_BF16_T   1
#define OUT_F32_BIAS 2

// ---- one-time f32 -> bf16 conversion (pure bandwidth pass) -----------------
__global__ __launch_bounds__(256) void cvt_f32_to_bf16(const float* __restrict__ in,
                                                       bf16_t* __restrict__ out, long n) {
  const long i = ((long)blockIdx.x * blockDim.x + threadIdx.x) * 8;
  if (i >= n) return;
  const float4 a = *reinterpret_cast<const float4*>(in + i);
  const float4 b = *reinterpret_cast<const float4*>(in + i + 4);
  union { uint4 q; bf16_t h[8]; } u;
  u.h[0] = (bf16_t)a.x; u.h[1] = (bf16_t)a.y; u.h[2] = (bf16_t)a.z; u.h[3] = (bf16_t)a.w;
  u.h[4] = (bf16_t)b.x; u.h[5] = (bf16_t)b.y; u.h[6] = (bf16_t)b.z; u.h[7] = (bf16_t)b.w;
  *reinterpret_cast<uint4*>(out + i) = u.q;
}

// ---- fragment loader -------------------------------------------------------
// 16-bit A-matrix 16x32 layout (ISA 7.12.2): lane L -> row M = L%16,
// K runs {kb..kb+7} and {16+kb..16+kb+7} with kb = (L/16)*8.
// B fragments use the mirrored layout (lane = N); every B operand here is a
// row-major "one row per output column" matrix, so both loaders are identical:
// two 16-byte (b128) loads per fragment per lane.
__device__ __forceinline__ v16bf load_frag(const bf16_t* __restrict__ base, long ld) {
  const int lane = threadIdx.x & 31;
  const int r    = lane & 15;
  const int kb   = (lane >> 4) << 3;
  const bf16_t* p = base + (long)r * ld + kb;
  union { v16bf v; uint4 q[2]; } u;
  u.q[0] = *reinterpret_cast<const uint4*>(p);
  u.q[1] = *reinterpret_cast<const uint4*>(p + 16);
  return u.v;
}

// ---- WMMA GEMM: C[M,N] = A[M,K] * B[N,K]^T (A,B bf16) ----------------------
// One wave computes a 64x64 tile; block = 8 waves; grid-stride over tiles.
template<int OMODE>
__global__ __launch_bounds__(256) void gemm_wmma(
    const bf16_t* __restrict__ Ap, long lda,
    const bf16_t* __restrict__ Bp, long ldb,
    void* __restrict__ Cp, long ldc,
    const float* __restrict__ bias,
    int M, int N, int K, float scale,
    int batchRows, long outBatchStride)
{
  const int lane   = threadIdx.x & 31;
  const int wave   = threadIdx.x >> 5;
  const int wpb    = blockDim.x >> 5;
  const int tilesN = N >> 6;
  const int nTiles = (M >> 6) * tilesN;

  for (int tile = blockIdx.x * wpb + wave; tile < nTiles; tile += (int)gridDim.x * wpb) {
    const int tm = (tile / tilesN) << 6;
    const int tn = (tile % tilesN) << 6;

    v8f zero = {};
    v8f acc[4][4];
#pragma unroll
    for (int i = 0; i < 4; ++i)
#pragma unroll
      for (int j = 0; j < 4; ++j) acc[i][j] = zero;

    const bf16_t* Arow = Ap + (long)tm * lda;
    const bf16_t* Brow = Bp + (long)tn * ldb;

    for (int k0 = 0; k0 < K; k0 += 32) {
      v16bf af[4], bfv[4];
#pragma unroll
      for (int i = 0; i < 4; ++i) af[i]  = load_frag(Arow + (long)(16 * i) * lda + k0, lda);
#pragma unroll
      for (int j = 0; j < 4; ++j) bfv[j] = load_frag(Brow + (long)(16 * j) * ldb + k0, ldb);
#pragma unroll
      for (int i = 0; i < 4; ++i)
#pragma unroll
        for (int j = 0; j < 4; ++j)
          acc[i][j] = __builtin_amdgcn_wmma_f32_16x16x32_bf16(
              false, af[i], false, bfv[j], (short)0, acc[i][j], false, false);
    }

    // D layout (ISA 7.12.2): lane -> col N = lane%16, VGPR r -> row M = r + 8*(lane/16)
    const int colLane = lane & 15;
    const int rowHalf = (lane >> 4) << 3;
#pragma unroll
    for (int i = 0; i < 4; ++i) {
#pragma unroll
      for (int j = 0; j < 4; ++j) {
        const int col     = tn + 16 * j + colLane;
        const int rowBase = tm + 16 * i + rowHalf;
        if constexpr (OMODE == OUT_BF16) {
          bf16_t* out = (bf16_t*)Cp;
#pragma unroll
          for (int r = 0; r < 8; ++r)
            out[(long)(rowBase + r) * ldc + col] = (bf16_t)(acc[i][j][r] * scale);
        } else if constexpr (OMODE == OUT_BF16_T) {
          // write D transposed, per batch: out[batch][col][row]  (packed b128 store)
          const int batch = rowBase / batchRows;
          const int rloc  = rowBase - batch * batchRows;
          bf16_t* out = (bf16_t*)Cp + (long)batch * outBatchStride + (long)col * batchRows + rloc;
          union { uint4 q; bf16_t h[8]; } u;
#pragma unroll
          for (int r = 0; r < 8; ++r) u.h[r] = (bf16_t)(acc[i][j][r] * scale);
          *reinterpret_cast<uint4*>(out) = u.q;
        } else {
          float* out = (float*)Cp;
          const float bv = bias[col];
#pragma unroll
          for (int r = 0; r < 8; ++r)
            out[(long)(rowBase + r) * ldc + col] = acc[i][j][r] * scale + bv;
        }
      }
    }
  }
}

// ---- in-place row softmax over 4096 bf16 elements --------------------------
__global__ __launch_bounds__(256) void softmax_rows_4096(bf16_t* __restrict__ S) {
  const int tid = threadIdx.x;
  bf16_t* p = S + (long)blockIdx.x * 4096;

  float v[16];
  float mx = -3.402823466e38f;
#pragma unroll
  for (int i = 0; i < 16; ++i) {
    v[i] = (float)p[tid + (i << 8)];
    mx = fmaxf(mx, v[i]);
  }
  __shared__ float red[256];
  red[tid] = mx;
  __syncthreads();
  for (int s = 128; s > 0; s >>= 1) {
    if (tid < s) red[tid] = fmaxf(red[tid], red[tid + s]);
    __syncthreads();
  }
  mx = red[0];
  __syncthreads();

  float sum = 0.f;
#pragma unroll
  for (int i = 0; i < 16; ++i) { v[i] = __expf(v[i] - mx); sum += v[i]; }
  red[tid] = sum;
  __syncthreads();
  for (int s = 128; s > 0; s >>= 1) {
    if (tid < s) red[tid] += red[tid + s];
    __syncthreads();
  }
  const float inv = 1.0f / red[0];
#pragma unroll
  for (int i = 0; i < 16; ++i) p[tid + (i << 8)] = (bf16_t)(v[i] * inv);
}

// ---- driver ----------------------------------------------------------------
extern "C" void kernel_launch(void* const* d_in, const int* in_sizes, int n_in,
                              void* d_out, int out_size, void* d_ws, size_t ws_size,
                              hipStream_t stream) {
  const float* x  = (const float*)d_in[0];
  const float* y  = (const float*)d_in[1];
  const float* Wq = (const float*)d_in[2];
  const float* Wk = (const float*)d_in[3];
  const float* Wv = (const float*)d_in[4];
  const float* Wo = (const float*)d_in[5];
  const float* bo = (const float*)d_in[6];

  constexpr int  B = 4, N = 4096, D = 1024, E = 1024;
  constexpr long BND = (long)B * N * D;   // x/y elements
  constexpr long BNE = (long)B * N * E;
  constexpr long WSZ = (long)E * D;       // weight elements

  // workspace (all bf16): xb | yb | Wq..Wo | q | k | vT | ctx | S(one batch)
  bf16_t* xb  = (bf16_t*)d_ws;
  bf16_t* yb  = xb  + BND;
  bf16_t* Wqb = yb  + BND;
  bf16_t* Wkb = Wqb + WSZ;
  bf16_t* Wvb = Wkb + WSZ;
  bf16_t* Wob = Wvb + WSZ;
  bf16_t* qb  = Wob + WSZ;
  bf16_t* kb  = qb  + BNE;
  bf16_t* vT  = kb  + BNE;       // [B][E][N]
  bf16_t* ctx = vT  + BNE;       // [B][N][E]
  bf16_t* Sm  = ctx + BNE;       // [N][N], reused per batch

  const dim3 blk(256);
  const int  M   = B * N;
  const float isq = 0.03125f;    // 1/sqrt(1024)

  // one-time conversions
  cvt_f32_to_bf16<<<(int)(BND / (256 * 8)), blk, 0, stream>>>(x,  xb,  BND);
  cvt_f32_to_bf16<<<(int)(BND / (256 * 8)), blk, 0, stream>>>(y,  yb,  BND);
  cvt_f32_to_bf16<<<(int)(WSZ / (256 * 8)), blk, 0, stream>>>(Wq, Wqb, WSZ);
  cvt_f32_to_bf16<<<(int)(WSZ / (256 * 8)), blk, 0, stream>>>(Wk, Wkb, WSZ);
  cvt_f32_to_bf16<<<(int)(WSZ / (256 * 8)), blk, 0, stream>>>(Wv, Wvb, WSZ);
  cvt_f32_to_bf16<<<(int)(WSZ / (256 * 8)), blk, 0, stream>>>(Wo, Wob, WSZ);

  // q = y*Wq^T, k = x*Wk^T (bf16 row-major), v = x*Wv^T stored transposed per batch
  gemm_wmma<OUT_BF16  ><<<(M / 64) * (E / 64) / 8, blk, 0, stream>>>(
      yb, D, Wqb, D, qb, E, nullptr, M, E, D, 1.f, 0, 0);
  gemm_wmma<OUT_BF16  ><<<(M / 64) * (E / 64) / 8, blk, 0, stream>>>(
      xb, D, Wkb, D, kb, E, nullptr, M, E, D, 1.f, 0, 0);
  gemm_wmma<OUT_BF16_T><<<(M / 64) * (E / 64) / 8, blk, 0, stream>>>(
      xb, D, Wvb, D, vT, 0, nullptr, M, E, D, 1.f, N, (long)E * N);

  for (int b = 0; b < B; ++b) {
    const bf16_t* qbb = qb  + (long)b * N * E;
    const bf16_t* kbb = kb  + (long)b * N * E;
    const bf16_t* vTb = vT  + (long)b * E * N;
    bf16_t*      ctxb = ctx + (long)b * N * E;

    // S = (q*k^T) / 32
    gemm_wmma<OUT_BF16><<<(N / 64) * (N / 64) / 8, blk, 0, stream>>>(
        qbb, E, kbb, E, Sm, N, nullptr, N, N, E, isq, 0, 0);
    // P = softmax_rows(S), in place
    softmax_rows_4096<<<N, blk, 0, stream>>>(Sm);
    // ctx = P * V  (B fragments from vT rows, contiguous)
    gemm_wmma<OUT_BF16><<<(N / 64) * (E / 64) / 8, blk, 0, stream>>>(
        Sm, N, vTb, N, ctxb, E, nullptr, N, E, N, 1.f, 0, 0);
  }

  // out = ctx*Wo^T + bo, f32
  gemm_wmma<OUT_F32_BIAS><<<(M / 64) * (D / 64) / 8, blk, 0, stream>>>(
      ctx, E, Wob, D, d_out, D, bo, M, D, E, 1.f, 0, 0);

  (void)in_sizes; (void)n_in; (void)out_size; (void)ws_size;
}
